// Network_20521353740407
// MI455X (gfx1250) — matrix operand order repaired and checked by reading, compiled, not run on hardware
//
#include <hip/hip_runtime.h>

typedef __attribute__((ext_vector_type(16))) _Float16 v16h;
typedef __attribute__((ext_vector_type(8)))  _Float16 v8h;
typedef __attribute__((ext_vector_type(8)))  float    v8f;

#define B_     32
#define NIN_   2312
#define KPAD_  2336      // 73 K-tiles of 32
#define NHID_  512
#define NOUT_  10
#define M2PAD_ 16
#define T_     350
#define TX_    384       // layer-1 padded time: 3 WGs x 128 -> zero guards in GEMM1
#define TPAD_  352       // layer-2 padded time: 22 N-tiles of 16

// SLAYER neuron constants
#define THETA_ 10.0f
#define DSR_   0.9048374180359595f     // exp(-1/tauSr)
#define ESR_   0.27182818284590452f    // e/tauSr  (alpha-kernel gain)
#define DREF_  0.36787944117144233f    // exp(-1/tauRef)
#define CREF_  (-54.365636569180904f)  // -scaleRef*theta*e

// workspace layout (bytes)
static constexpr size_t OFF_W1H = 0;                                   // 512*2336*2
static constexpr size_t OFF_W2H = OFF_W1H + (size_t)NHID_*KPAD_*2;     // 16*512*2
static constexpr size_t OFF_XHT = OFF_W2H + (size_t)M2PAD_*NHID_*2;    // B*TX*KPAD*2
static constexpr size_t OFF_U1  = OFF_XHT + (size_t)B_*TX_*KPAD_*2;    // B*NHID*TX*4
static constexpr size_t OFF_P1T = OFF_U1  + (size_t)B_*NHID_*TX_*4;    // B*TPAD*NHID*2
static constexpr size_t OFF_U2  = OFF_P1T + (size_t)B_*TPAD_*NHID_*2;  // B*16*TPAD*4

// gfx1250 async global->LDS staging (ASYNCcnt-tracked, no VGPR round trip).
// Inline asm per CDNA5 ISA; LDS byte address = low 32 bits
// of the flat pointer (ISA 10.2).
#define USE_ASYNC_LDS 1

__device__ inline void async_copy_b128(const _Float16* g, _Float16* l) {
  unsigned loff = (unsigned)(uintptr_t)l;
  asm volatile("global_load_async_to_lds_b128 %0, %1, off"
               :: "v"(loff), "v"(g) : "memory");
}

__device__ inline void wait_asynccnt0() {
#if defined(__has_builtin) && __has_builtin(__builtin_amdgcn_s_wait_asynccnt)
  __builtin_amdgcn_s_wait_asynccnt(0);
#else
  asm volatile("s_wait_asynccnt 0x0" ::: "memory");
#endif
}

__device__ inline v16h load_frag(const _Float16* p0, const _Float16* p1) {
  v8h lo = *(const v8h*)p0;
  v8h hi = *(const v8h*)p1;
  return __builtin_shufflevector(lo, hi, 0,1,2,3,4,5,6,7,8,9,10,11,12,13,14,15);
}

// ---- staging / conversion kernels -------------------------------------------
__global__ void conv_w1(const float* __restrict__ W1, _Float16* __restrict__ w1h) {
  int idx = blockIdx.x * blockDim.x + threadIdx.x;
  if (idx >= NHID_ * KPAD_) return;
  int m = idx / KPAD_, k = idx % KPAD_;
  w1h[idx] = (k < NIN_) ? (_Float16)W1[(size_t)m * NIN_ + k] : (_Float16)0.0f;
}

__global__ void conv_w2(const float* __restrict__ W2, _Float16* __restrict__ w2h) {
  int idx = blockIdx.x * blockDim.x + threadIdx.x;
  if (idx >= M2PAD_ * NHID_) return;
  int o = idx / NHID_, k = idx % NHID_;
  w2h[idx] = (o < NOUT_) ? (_Float16)W2[(size_t)o * NHID_ + k] : (_Float16)0.0f;
}

// time-major f16 transpose of the spike raster: xht[b][t][k] = X[b][k][t]
__global__ void conv_x(const float* __restrict__ X, _Float16* __restrict__ xht) {
  size_t idx = (size_t)blockIdx.x * blockDim.x + threadIdx.x;
  if (idx >= (size_t)B_ * TX_ * KPAD_) return;
  int k = (int)(idx % KPAD_);
  int t = (int)((idx / KPAD_) % TX_);
  int b = (int)(idx / ((size_t)KPAD_ * TX_));
  xht[idx] = (k < NIN_ && t < T_) ? (_Float16)X[((size_t)b * NIN_ + k) * T_ + t]
                                  : (_Float16)0.0f;
}

// ---- GEMM1: U1[b,h,t] = sum_k W1[h,k] * X[b,k,t]  (f16 WMMA, f32 accum) -----
// WG = 4 waves. WG tile 64(M) x 128(N); A tile double-buffered in LDS via
// async global->LDS loads; each wave owns 32 N-columns (2 WMMA col-tiles).
__global__ __launch_bounds__(128) void gemm1_wmma(
    const _Float16* __restrict__ w1h,   // [NHID][KPAD]
    const _Float16* __restrict__ xht,   // [B][TX][KPAD]
    float* __restrict__ u1)             // [B][NHID][TX]
{
  __shared__ _Float16 lds_a[2][64 * 32];
  const int tid  = threadIdx.x;
  const int lane = tid & 31;
  const int wave = tid >> 5;
  const int m0   = blockIdx.x * 64;
  const int n0w  = blockIdx.y * 128 + wave * 32;
  const int b    = blockIdx.z;
  const int lrow = lane & 15;
  const int hi   = lane >> 4;                 // 0|1 half-wave split
  const int ac0  = hi ? 8 : 0;

  // staging map: each thread moves 2 b128 chunks of the 64x32 A tile
  const int c0   = tid * 2;
  const int row0 = c0 >> 2,       ch0 = (c0 & 3) * 8;
  const int row1 = (c0 + 1) >> 2, ch1 = ((c0 + 1) & 3) * 8;

  auto stage = [&](int k0, int buf) {
    const _Float16* g0 = &w1h[(size_t)(m0 + row0) * KPAD_ + k0 + ch0];
    const _Float16* g1 = &w1h[(size_t)(m0 + row1) * KPAD_ + k0 + ch1];
    _Float16* l0 = &lds_a[buf][row0 * 32 + ch0];
    _Float16* l1 = &lds_a[buf][row1 * 32 + ch1];
#if USE_ASYNC_LDS
    async_copy_b128(g0, l0);
    async_copy_b128(g1, l1);
#else
    *(uint4*)l0 = *(const uint4*)g0;
    *(uint4*)l1 = *(const uint4*)g1;
#endif
  };

  v8f acc[4][2] = {};
  const _Float16* bptr0 = xht + ((size_t)b * TX_ + n0w + lrow) * KPAD_ + (hi ? 8 : 0);
  const _Float16* bptr1 = bptr0 + (size_t)16 * KPAD_;

  stage(0, 0);
#if USE_ASYNC_LDS
  wait_asynccnt0();
#endif
  __syncthreads();

  const int KT = KPAD_ / 32;   // 73
  for (int kt = 0; kt < KT; ++kt) {
    const int buf = kt & 1;
    if (kt + 1 < KT) stage((kt + 1) * 32, buf ^ 1);   // prefetch next A tile

    v16h afr[4];
    #pragma unroll
    for (int mt = 0; mt < 4; ++mt) {
      const _Float16* ap = &lds_a[buf][(mt * 16 + lrow) * 32 + ac0];
      afr[mt] = load_frag(ap, ap + 16);     // K{0..7,16..23} | K{8..15,24..31}
    }

    const _Float16* bp0 = bptr0 + (size_t)kt * 32;
    const _Float16* bp1 = bptr1 + (size_t)kt * 32;
    __builtin_prefetch((const void*)(bp0 + 64), 0, 0);  // global_prefetch_b8
    v16h b0 = load_frag(bp0, bp0 + 16);
    v16h b1 = load_frag(bp1, bp1 + 16);

    #pragma unroll
    for (int mt = 0; mt < 4; ++mt)
      acc[mt][0] = __builtin_amdgcn_wmma_f32_16x16x32_f16(
          false, afr[mt], false, b0, (short)0, acc[mt][0], false, false);
    #pragma unroll
    for (int mt = 0; mt < 4; ++mt)
      acc[mt][1] = __builtin_amdgcn_wmma_f32_16x16x32_f16(
          false, afr[mt], false, b1, (short)0, acc[mt][1], false, false);

#if USE_ASYNC_LDS
    wait_asynccnt0();                       // next-tile LDS writes landed
#endif
    __syncthreads();                        // single barrier per iteration
  }

  #pragma unroll
  for (int nt = 0; nt < 2; ++nt) {
    int col = n0w + nt * 16 + lrow;
    #pragma unroll
    for (int mt = 0; mt < 4; ++mt)
      #pragma unroll
      for (int j = 0; j < 8; ++j) {
        int row = m0 + mt * 16 + j + hi * 8;  // C/D layout: VGPR j -> M=j / j+8
        u1[((size_t)b * NHID_ + row) * TX_ + col] = acc[mt][nt][j];
      }
  }
}

// ---- fused scan 1: psp-IIR -> membrane+refractory -> spike -> psp(s1) -------
__global__ __launch_bounds__(256) void scan1(
    const float* __restrict__ u1,   // [B][NHID][TX]
    _Float16* __restrict__ p1t)     // [B][TPAD][NHID] (time-major for GEMM2)
{
  int idx = blockIdx.x * blockDim.x + threadIdx.x;
  if (idx >= B_ * NHID_) return;
  int b = idx / NHID_, h = idx % NHID_;
  const float* up = u1 + (size_t)idx * TX_;
  _Float16* pp = p1t + (size_t)b * TPAD_ * NHID_ + h;
  float pa = 0.f, pb = 0.f, ra = 0.f, rb = 0.f, qa = 0.f, qb = 0.f;
  for (int t = 0; t < T_; ++t) {
    float g = up[t];
    pb = DSR_ * (pb + pa);            // alpha-IIR of dense drive (uses old pa)
    pa = DSR_ * pa + g;
    float u = ESR_ * pb + CREF_ * rb; // membrane = psp + refractory
    float s = (u >= THETA_) ? 1.0f : 0.0f;
    rb = DREF_ * (rb + ra);           // refractory alpha-IIR (old ra)
    ra = DREF_ * ra + s;
    qb = DSR_ * (qb + qa);            // fused psp(s1) for layer 2
    qa = DSR_ * qa + s;
    pp[(size_t)t * NHID_] = (_Float16)(ESR_ * qb);
  }
  pp[(size_t)T_ * NHID_]       = (_Float16)0.0f;  // zero time padding
  pp[(size_t)(T_ + 1) * NHID_] = (_Float16)0.0f;
}

// ---- GEMM2: U2[b,o,t] = sum_k W2[o,k] * P1[b,k,t], one 16x16 tile per wave --
__global__ __launch_bounds__(256) void gemm2_wmma(
    const _Float16* __restrict__ w2h,   // [16][512]
    const _Float16* __restrict__ p1t,   // [B][TPAD][NHID]
    float* __restrict__ u2)             // [B][16][TPAD]
{
  const int tid  = threadIdx.x;
  const int lane = tid & 31;
  const int wg   = blockIdx.x * 8 + (tid >> 5);
  const int NT   = TPAD_ / 16;          // 22
  if (wg >= B_ * NT) return;            // wave-uniform
  const int b = wg / NT, nt = wg % NT;
  const int lrow = lane & 15;
  const int hi   = lane >> 4;
  v8f acc = {};
  const _Float16* abase = w2h + (size_t)lrow * NHID_ + hi * 8;
  const _Float16* bbase =
      p1t + ((size_t)b * TPAD_ + nt * 16 + lrow) * NHID_ + hi * 8;
  #pragma unroll 4
  for (int k0 = 0; k0 < NHID_; k0 += 32) {
    v16h a  = load_frag(abase + k0, abase + k0 + 16);
    v16h bb = load_frag(bbase + k0, bbase + k0 + 16);
    acc = __builtin_amdgcn_wmma_f32_16x16x32_f16(
        false, a, false, bb, (short)0, acc, false, false);
  }
  #pragma unroll
  for (int j = 0; j < 8; ++j)
    u2[((size_t)b * M2PAD_ + j + hi * 8) * TPAD_ + nt * 16 + lrow] = acc[j];
}

// ---- scan 2: refractory + threshold on output layer -------------------------
__global__ void scan2(const float* __restrict__ u2, float* __restrict__ out) {
  int idx = blockIdx.x * blockDim.x + threadIdx.x;
  if (idx >= B_ * M2PAD_) return;
  int b = idx / M2PAD_, o = idx % M2PAD_;
  const float* up = u2 + (size_t)idx * TPAD_;
  float ra = 0.f, rb = 0.f;
  for (int t = 0; t < T_; ++t) {
    float u = up[t] + CREF_ * rb;
    float s = (u >= THETA_) ? 1.0f : 0.0f;
    rb = DREF_ * (rb + ra);
    ra = DREF_ * ra + s;
    if (o < NOUT_) out[((size_t)b * NOUT_ + o) * T_ + t] = s;
  }
}

extern "C" void kernel_launch(void* const* d_in, const int* in_sizes, int n_in,
                              void* d_out, int out_size, void* d_ws, size_t ws_size,
                              hipStream_t stream) {
  (void)in_sizes; (void)n_in; (void)out_size; (void)ws_size;
  const float* X  = (const float*)d_in[0];
  const float* W1 = (const float*)d_in[1];
  const float* W2 = (const float*)d_in[2];
  char* ws = (char*)d_ws;
  _Float16* w1h = (_Float16*)(ws + OFF_W1H);
  _Float16* w2h = (_Float16*)(ws + OFF_W2H);
  _Float16* xht = (_Float16*)(ws + OFF_XHT);
  float*    u1  = (float*)   (ws + OFF_U1);
  _Float16* p1t = (_Float16*)(ws + OFF_P1T);
  float*    u2  = (float*)   (ws + OFF_U2);

  conv_w1<<<(NHID_ * KPAD_ + 255) / 256, 256, 0, stream>>>(W1, w1h);
  conv_w2<<<(M2PAD_ * NHID_ + 255) / 256, 256, 0, stream>>>(W2, w2h);
  size_t nx = (size_t)B_ * TX_ * KPAD_;
  conv_x<<<(unsigned)((nx + 255) / 256), 256, 0, stream>>>(X, xht);

  dim3 g1(NHID_ / 64, TX_ / 128, B_);
  gemm1_wmma<<<g1, 128, 0, stream>>>(w1h, xht, u1);

  scan1<<<(B_ * NHID_ + 255) / 256, 256, 0, stream>>>(u1, p1t);

  gemm2_wmma<<<(B_ * (TPAD_ / 16) + 7) / 8, 256, 0, stream>>>(w2h, p1t, u2);

  scan2<<<(B_ * M2PAD_ + 255) / 256, 256, 0, stream>>>(u2, (float*)d_out);
}
